// hyperBlock_36928128811139
// MI455X (gfx1250) — matrix operand order repaired and checked by reading, not compiled, not run on hardware
//
#include <hip/hip_runtime.h>
#include <hip/hip_bf16.h>
#include <math.h>

#define B_    16
#define DIM_  512
#define H_    56
#define W_    56
#define P_    (H_*W_)      // 3136
#define STYLE_ 64
#define LAT_  8
#define K_    7
#define HID_  2048
#define EPS_  1e-6f

typedef __attribute__((ext_vector_type(16))) __bf16        v16bf;
typedef __attribute__((ext_vector_type(8)))  float         v8f;
typedef __attribute__((ext_vector_type(4)))  unsigned int  u32x4;
typedef __attribute__((ext_vector_type(8)))  int           i32x8;
typedef __attribute__((ext_vector_type(4)))  int           i32x4;

union FragBF { v16bf v; u32x4 q[2]; };

__device__ __forceinline__ unsigned short f2bf(float f) {
  unsigned int u = __float_as_uint(f);
  u += 0x7FFFu + ((u >> 16) & 1u);          // round-to-nearest-even
  return (unsigned short)(u >> 16);
}

// ---------------------------------------------------------------------------
// TDM: DMA a 2-D bf16 tile (tile1 rows x tile0 elems, row stride stride0)
// from global memory to a contiguous LDS region. Tracked by TENSORcnt.
// ---------------------------------------------------------------------------
__device__ __forceinline__ void tdm_load_2d(unsigned lds_off, const void* gptr,
                                            unsigned tile0, unsigned tile1,
                                            unsigned tdim0, unsigned tdim1,
                                            unsigned stride0) {
  const unsigned long long ga = (unsigned long long)(size_t)gptr;
  u32x4 g0;
  g0[0] = 1u;                                              // count=1, user D#
  g0[1] = lds_off;                                         // LDS byte address
  g0[2] = (unsigned)ga;                                    // global_addr[31:0]
  g0[3] = (unsigned)((ga >> 32) & 0x1FFFFFFull) | (2u << 30);  // [56:32] | type=2
  i32x8 g1;
  g1[0] = (int)(1u << 16);                                 // data_size=1 (2B)
  g1[1] = (int)((tdim0 & 0xFFFFu) << 16);                  // tensor_dim0 lo16
  g1[2] = (int)((tdim0 >> 16) | ((tdim1 & 0xFFFFu) << 16));// dim0 hi / dim1 lo
  g1[3] = (int)((tdim1 >> 16) | (tile0 << 16));            // dim1 hi / tile_dim0
  g1[4] = (int)(tile1 & 0xFFFFu);                          // tile_dim1 (tile2=0)
  g1[5] = (int)stride0;                                    // dim0_stride[31:0]
  g1[6] = 0;                                               // stride hi / dim1_stride
  g1[7] = 0;
  i32x4 gz = {0, 0, 0, 0};
#if defined(__clang_major__) && (__clang_major__ >= 23)
  i32x8 gz8 = {0, 0, 0, 0, 0, 0, 0, 0};
  __builtin_amdgcn_tensor_load_to_lds(g0, g1, gz, gz, gz8, 0);
#else
  __builtin_amdgcn_tensor_load_to_lds(g0, g1, gz, gz, 0);
#endif
}

// ---------------------------------------------------------------------------
// 1) style -> latent z vectors (3x) and per-sample biases b_dw, b1, b2
// ---------------------------------------------------------------------------
__global__ void k_gen_zb(const float* __restrict__ s,
    const float* __restrict__ dw_fc_w,  const float* __restrict__ dw_fc_b,
    const float* __restrict__ pw1_fc_w, const float* __restrict__ pw1_fc_b,
    const float* __restrict__ pw2_fc_w, const float* __restrict__ pw2_fc_b,
    const float* __restrict__ dw_bfc_w, const float* __restrict__ dw_bfc_b,
    const float* __restrict__ pw1_bfc_w,const float* __restrict__ pw1_bfc_b,
    const float* __restrict__ pw2_bfc_w,const float* __restrict__ pw2_bfc_b,
    float* __restrict__ z_dw, float* __restrict__ z1, float* __restrict__ z2,
    float* __restrict__ b_dw, float* __restrict__ b1, float* __restrict__ b2) {
  const int b = blockIdx.x, t = threadIdx.x;
  const float* sb = s + b * STYLE_;
  if (t < 3 * LAT_) {
    const int which = t / LAT_, l = t % LAT_;
    const float* fw = which == 0 ? dw_fc_w : which == 1 ? pw1_fc_w : pw2_fc_w;
    const float* fb = which == 0 ? dw_fc_b : which == 1 ? pw1_fc_b : pw2_fc_b;
    float a = fb[l];
    for (int j = 0; j < STYLE_; ++j) a = fmaf(sb[j], fw[j * LAT_ + l], a);
    float* zo = which == 0 ? z_dw : which == 1 ? z1 : z2;
    zo[b * LAT_ + l] = a;
  }
  for (int c = t; c < HID_; c += 256) {
    float a = pw1_bfc_b[c];
    for (int j = 0; j < STYLE_; ++j) a = fmaf(sb[j], pw1_bfc_w[j * HID_ + c], a);
    b1[b * HID_ + c] = a;
  }
  for (int c = t; c < DIM_; c += 256) {
    float a  = dw_bfc_b[c];
    float a2 = pw2_bfc_b[c];
    for (int j = 0; j < STYLE_; ++j) {
      a  = fmaf(sb[j], dw_bfc_w[j * DIM_ + c],  a);
      a2 = fmaf(sb[j], pw2_bfc_w[j * DIM_ + c], a2);
    }
    b_dw[b * DIM_ + c] = a;
    b2[b * DIM_ + c]   = a2;
  }
}

// ---------------------------------------------------------------------------
// 2) per-sample depthwise 7x7 kernels
// ---------------------------------------------------------------------------
__global__ void k_gen_wdw(const float* __restrict__ z_dw,
                          const float* __restrict__ dw_param,
                          float* __restrict__ w_dw) {
  const int idx = blockIdx.x * 256 + threadIdx.x;       // < B*DIM*49
  const int b   = idx / (DIM_ * 49);
  const int rem = idx - b * (DIM_ * 49);
  float a = 0.f;
#pragma unroll
  for (int l = 0; l < LAT_; ++l)
    a = fmaf(z_dw[b * LAT_ + l], dw_param[(size_t)l * (DIM_ * 49) + rem], a);
  w_dw[idx] = a;
}

// ---------------------------------------------------------------------------
// 3) generated 1x1 weights, transposed + bf16: out[b, inner, o] (bf16)
//    param layout [LAT, Odim, Inner]
// ---------------------------------------------------------------------------
__global__ void k_gen_wt(const float* __restrict__ z,
                         const float* __restrict__ param,
                         unsigned short* __restrict__ out,
                         int Odim, int Inner) {
  const size_t idx = (size_t)blockIdx.x * 256 + threadIdx.x;  // < B*Inner*Odim
  const int per   = Inner * Odim;
  const int b     = (int)(idx / per);
  const int rem   = (int)(idx - (size_t)b * per);
  const int inner = rem / Odim, o = rem - inner * Odim;
  float a = 0.f;
#pragma unroll
  for (int l = 0; l < LAT_; ++l)
    a = fmaf(z[b * LAT_ + l], param[((size_t)l * Odim + o) * Inner + inner], a);
  out[idx] = f2bf(a);
}

// ---------------------------------------------------------------------------
// 4) fused reflect-pad depthwise conv + bias + channels-first LayerNorm.
//    Output pixel-major bf16: Xt[b, p, c]  (WMMA A-matrix source)
// ---------------------------------------------------------------------------
__global__ void __launch_bounds__(128) k_dwln(
    const float* __restrict__ x, const float* __restrict__ w_dw,
    const float* __restrict__ b_dw, const float* __restrict__ ln_w,
    const float* __restrict__ ln_b, unsigned short* __restrict__ Xt) {
  const int p = blockIdx.x;           // 0..P_-1
  const int b = blockIdx.y;
  const int t = threadIdx.x;          // 0..127
  const int oh = p / W_, ow = p - oh * W_;
  int rs[K_], cs[K_];
#pragma unroll
  for (int i = 0; i < K_; ++i) {
    int r = oh - 3 + i; r = r < 0 ? -r : r; r = r >= H_ ? 2 * H_ - 2 - r : r;
    int c = ow - 3 + i; c = c < 0 ? -c : c; c = c >= W_ ? 2 * W_ - 2 - c : c;
    rs[i] = r; cs[i] = c;
  }
  float vals[4];
  float ls = 0.f, lsq = 0.f;
#pragma unroll
  for (int q = 0; q < 4; ++q) {
    const int c = t + q * 128;
    const float* xb = x    + ((size_t)(b * DIM_ + c)) * P_;
    const float* wb = w_dw + ((size_t)(b * DIM_ + c)) * 49;
    float a = b_dw[b * DIM_ + c];
#pragma unroll
    for (int i = 0; i < K_; ++i)
#pragma unroll
      for (int j = 0; j < K_; ++j)
        a = fmaf(xb[rs[i] * W_ + cs[j]], wb[i * K_ + j], a);
    vals[q] = a; ls += a; lsq += a * a;
  }
  __shared__ float s1[128], s2[128];
  s1[t] = ls; s2[t] = lsq;
  __syncthreads();
  for (int w = 64; w > 0; w >>= 1) {
    if (t < w) { s1[t] += s1[t + w]; s2[t] += s2[t + w]; }
    __syncthreads();
  }
  const float mean = s1[0] * (1.f / DIM_);
  const float var  = s2[0] * (1.f / DIM_) - mean * mean;
  const float rstd = rsqrtf(var + EPS_);
  unsigned short* dst = Xt + ((size_t)b * P_ + p) * DIM_;
#pragma unroll
  for (int q = 0; q < 4; ++q) {
    const int c = t + q * 128;
    dst[c] = f2bf((vals[q] - mean) * rstd * ln_w[c] + ln_b[c]);
  }
}

// ---------------------------------------------------------------------------
// Shared GEMM mainloop. Block = 256 threads = 8 waves.
// Block tile: 32 pixels x 256 out-channels. Wave tile: 16 x 64 (4 wmma accs).
// A (32x32 bf16) and B (32x256 bf16) double-buffered in LDS.
// Tiles are DMA'd by the Tensor Data Mover (wave 0 issues, TENSORcnt-tracked),
// overlapping with WMMA compute on the other buffer; one barrier per K-step.
// ---------------------------------------------------------------------------
template <int CIN, int COUT>
__device__ __forceinline__ void gemm_mainloop(
    const unsigned short* __restrict__ Xb,   // [P_, CIN] bf16
    const unsigned short* __restrict__ Wb,   // [CIN, COUT] bf16
    int p0, int n0, unsigned short* lA, unsigned short* lB, v8f acc[4]) {
  const int t    = threadIdx.x;
  const int lane = t & 31, wave = t >> 5;
  const int wr = wave >> 2, wc = wave & 3;   // wave row (pixels), wave col (N)
  const int m = lane & 15, hi = lane >> 4;

  // Flat LDS aperture keeps the LDS byte offset in addr[31:0].
  const unsigned ldsA = (unsigned)(size_t)lA;
  const unsigned ldsB = (unsigned)(size_t)lB;

  auto issueTDM = [&](int k0, int buf) {
    // A tile: 32 pixel-rows x 32 K halfs, row stride CIN
    tdm_load_2d(ldsA + (unsigned)buf * (32 * 32 * 2),
                Xb + (size_t)p0 * CIN + k0,
                /*tile0=*/32, /*tile1=*/32,
                /*tdim0=*/CIN, /*tdim1=*/P_, /*stride0=*/CIN);
    // B tile: 32 K-rows x 256 N halfs, row stride COUT
    tdm_load_2d(ldsB + (unsigned)buf * (32 * 256 * 2),
                Wb + (size_t)k0 * COUT + n0,
                /*tile0=*/256, /*tile1=*/32,
                /*tdim0=*/COUT, /*tdim1=*/CIN, /*stride0=*/COUT);
  };

  if (wave == 0) {
    issueTDM(0, 0);
    __builtin_amdgcn_s_wait_tensorcnt(0);
  }
  __syncthreads();

  int buf = 0;
  for (int k0 = 0; k0 < CIN; k0 += 32) {
    if (wave == 0 && k0 + 32 < CIN) issueTDM(k0 + 32, buf ^ 1);  // async DMA
    if (k0 + 64 < CIN)
      __builtin_prefetch(
          (const void*)(Wb + (size_t)(k0 + 64 + (t >> 3)) * COUT + n0 + (t & 7) * 32),
          0, 3);
    // A fragment: lane m holds row (wr*16+m); K chunks 0-7/16-23 (hi=0) or 8-15/24-31
    FragBF a;
    const unsigned short* as = lA + buf * (32 * 32) + (wr * 16 + m) * 32 + hi * 8;
    a.q[0] = *(const u32x4*)(as);
    a.q[1] = *(const u32x4*)(as + 16);
#pragma unroll
    for (int j = 0; j < 4; ++j) {
      // B fragment: lane holds column N=m; K chunks 0-7/16-23 (hi=0) or 8-15/24-31
      FragBF bfr;
      const unsigned short* bs = lB + buf * (32 * 256) + hi * 8 * 256 + wc * 64 + j * 16 + m;
#pragma unroll
      for (int s = 0; s < 16; ++s)
        bfr.v[s] = __builtin_bit_cast(__bf16, bs[((s & 7) + (s >> 3) * 16) * 256]);
      acc[j] = __builtin_amdgcn_wmma_f32_16x16x32_bf16(
          false, a.v, false, bfr.v, (short)0, acc[j], false, false);
    }
    if (wave == 0) __builtin_amdgcn_s_wait_tensorcnt(0);  // drain DMA of buf^1
    __syncthreads();                                      // publish buf^1
    buf ^= 1;
  }
}

// ---------------------------------------------------------------------------
// 5) GEMM1 + bias + exact GELU -> bf16 H1[b, p, o]
// ---------------------------------------------------------------------------
__global__ void __launch_bounds__(256) k_gemm1(
    const unsigned short* __restrict__ Xt, const unsigned short* __restrict__ Wt1,
    const float* __restrict__ b1, unsigned short* __restrict__ H1) {
  const int lane = threadIdx.x & 31, wave = threadIdx.x >> 5;
  const int wr = wave >> 2, wc = wave & 3;
  const int m = lane & 15, hi = lane >> 4;
  const int p0 = blockIdx.x * 32;
  const int n0 = blockIdx.y * 256;
  const int b  = blockIdx.z;
  __shared__ __align__(16) unsigned short lA[2 * 32 * 32];
  __shared__ __align__(16) unsigned short lB[2 * 32 * 256];
  v8f acc[4];
#pragma unroll
  for (int j = 0; j < 4; ++j) acc[j] = (v8f){0.f,0.f,0.f,0.f,0.f,0.f,0.f,0.f};
  gemm_mainloop<DIM_, HID_>(Xt + ((size_t)b * P_) * DIM_,
                            Wt1 + ((size_t)b * DIM_) * HID_, p0, n0, lA, lB, acc);
  const int mb = p0 + wr * 16 + hi * 8;
#pragma unroll
  for (int j = 0; j < 4; ++j) {
    const int   n  = n0 + wc * 64 + j * 16 + m;
    const float bn = b1[b * HID_ + n];
#pragma unroll
    for (int r = 0; r < 8; ++r) {
      float v = acc[j][r] + bn;
      float g = 0.5f * v * (1.f + erff(v * 0.70710678118f));  // exact GELU
      H1[((size_t)b * P_ + mb + r) * HID_ + n] = f2bf(g);
    }
  }
}

// ---------------------------------------------------------------------------
// 6) GEMM2 + bias + gamma + residual; scatter to channel-major f32 output.
// ---------------------------------------------------------------------------
__global__ void __launch_bounds__(256) k_gemm2(
    const unsigned short* __restrict__ Ht, const unsigned short* __restrict__ Wt2,
    const float* __restrict__ b2, const float* __restrict__ xin,
    const float* __restrict__ gamma_p, float* __restrict__ out) {
  const int lane = threadIdx.x & 31, wave = threadIdx.x >> 5;
  const int wr = wave >> 2, wc = wave & 3;
  const int m = lane & 15, hi = lane >> 4;
  const int p0 = blockIdx.x * 32;
  const int c0 = blockIdx.y * 256;
  const int b  = blockIdx.z;
  __shared__ __align__(16) unsigned short lA[2 * 32 * 32];
  __shared__ __align__(16) unsigned short lB[2 * 32 * 256];
  v8f acc[4];
#pragma unroll
  for (int j = 0; j < 4; ++j) acc[j] = (v8f){0.f,0.f,0.f,0.f,0.f,0.f,0.f,0.f};
  gemm_mainloop<HID_, DIM_>(Ht + ((size_t)b * P_) * HID_,
                            Wt2 + ((size_t)b * HID_) * DIM_, p0, c0, lA, lB, acc);
  const int mb = p0 + wr * 16 + hi * 8;
#pragma unroll
  for (int j = 0; j < 4; ++j) {
    const int   n  = c0 + wc * 64 + j * 16 + m;   // output channel
    const float g  = gamma_p[n];
    const float bn = b2[b * DIM_ + n];
    const size_t obase = ((size_t)b * DIM_ + n) * P_;
#pragma unroll
    for (int r = 0; r < 8; ++r) {
      const size_t o = obase + (size_t)(mb + r);
      out[o] = xin[o] + g * (acc[j][r] + bn);
    }
  }
}

// ---------------------------------------------------------------------------
extern "C" void kernel_launch(void* const* d_in, const int* in_sizes, int n_in,
                              void* d_out, int out_size, void* d_ws, size_t ws_size,
                              hipStream_t stream) {
  const float* x         = (const float*)d_in[0];
  const float* s         = (const float*)d_in[1];
  const float* dw_fc_w   = (const float*)d_in[2];
  const float* dw_fc_b   = (const float*)d_in[3];
  const float* dw_param  = (const float*)d_in[4];
  const float* dw_bfc_w  = (const float*)d_in[5];
  const float* dw_bfc_b  = (const float*)d_in[6];
  const float* ln_w      = (const float*)d_in[7];
  const float* ln_b      = (const float*)d_in[8];
  const float* pw1_fc_w  = (const float*)d_in[9];
  const float* pw1_fc_b  = (const float*)d_in[10];
  const float* pw1_param = (const float*)d_in[11];
  const float* pw1_bfc_w = (const float*)d_in[12];
  const float* pw1_bfc_b = (const float*)d_in[13];
  const float* pw2_fc_w  = (const float*)d_in[14];
  const float* pw2_fc_b  = (const float*)d_in[15];
  const float* pw2_param = (const float*)d_in[16];
  const float* pw2_bfc_w = (const float*)d_in[17];
  const float* pw2_bfc_b = (const float*)d_in[18];
  const float* gamma_p   = (const float*)d_in[19];
  float* out = (float*)d_out;

  char* w = (char*)d_ws;
  size_t off = 0;
  auto carve = [&](size_t bytes) -> void* {
    void* p = w + off;
    off += (bytes + 255) & ~(size_t)255;
    return p;
  };
  float* z_dw = (float*)carve(B_ * LAT_ * sizeof(float));
  float* z1   = (float*)carve(B_ * LAT_ * sizeof(float));
  float* z2   = (float*)carve(B_ * LAT_ * sizeof(float));
  float* b_dw = (float*)carve(B_ * DIM_ * sizeof(float));
  float* b1   = (float*)carve(B_ * HID_ * sizeof(float));
  float* b2   = (float*)carve(B_ * DIM_ * sizeof(float));
  float* w_dw = (float*)carve((size_t)B_ * DIM_ * 49 * sizeof(float));
  unsigned short* Xt  = (unsigned short*)carve((size_t)B_ * P_ * DIM_ * 2);
  unsigned short* Wt1 = (unsigned short*)carve((size_t)B_ * DIM_ * HID_ * 2);
  unsigned short* Wt2 = (unsigned short*)carve((size_t)B_ * HID_ * DIM_ * 2);
  unsigned short* H1  = (unsigned short*)carve((size_t)B_ * P_ * HID_ * 2);
  (void)ws_size; (void)in_sizes; (void)n_in; (void)out_size;

  // 1) z vectors + biases
  k_gen_zb<<<dim3(B_), dim3(256), 0, stream>>>(
      s, dw_fc_w, dw_fc_b, pw1_fc_w, pw1_fc_b, pw2_fc_w, pw2_fc_b,
      dw_bfc_w, dw_bfc_b, pw1_bfc_w, pw1_bfc_b, pw2_bfc_w, pw2_bfc_b,
      z_dw, z1, z2, b_dw, b1, b2);
  // 2) depthwise kernels
  k_gen_wdw<<<dim3((B_ * DIM_ * 49) / 256), dim3(256), 0, stream>>>(
      z_dw, dw_param, w_dw);
  // 3) generated 1x1 weights (transposed, bf16)
  k_gen_wt<<<dim3((unsigned)((size_t)B_ * DIM_ * HID_ / 256)), dim3(256), 0, stream>>>(
      z1, pw1_param, Wt1, HID_, DIM_);
  k_gen_wt<<<dim3((unsigned)((size_t)B_ * HID_ * DIM_ / 256)), dim3(256), 0, stream>>>(
      z2, pw2_param, Wt2, DIM_, HID_);
  // 4) depthwise conv + LN -> pixel-major bf16
  k_dwln<<<dim3(P_, B_), dim3(128), 0, stream>>>(x, w_dw, b_dw, ln_w, ln_b, Xt);
  // 5) GEMM1 + GELU -> bf16
  k_gemm1<<<dim3(P_ / 32, HID_ / 256, B_), dim3(256), 0, stream>>>(Xt, Wt1, b1, H1);
  // 6) GEMM2 + bias + gamma + residual -> f32 channel-major
  k_gemm2<<<dim3(P_ / 32, DIM_ / 256, B_), dim3(256), 0, stream>>>(
      H1, Wt2, b2, x, gamma_p, out);
}
